// Causal_attention_45938970198723
// MI455X (gfx1250) — matrix operand order repaired and checked by reading, not compiled, not run on hardware
//
#include <hip/hip_runtime.h>
#include <hip/hip_bf16.h>
#include <math.h>

typedef __attribute__((ext_vector_type(16))) _Float16 v16h;
typedef __attribute__((ext_vector_type(8)))  _Float16 v8h;
typedef __attribute__((ext_vector_type(8)))  float    v8f;

#define TK   32   // K-depth per LDS stage (one WMMA k-step)
#define LDH  40   // padded LDS row stride in halves (80B rows, 16B aligned)

// ---------------------------------------------------------------------------
// Fragment helpers (wave32 layouts per CDNA5 ISA §7.12.2)
//  A (16x32 f16): lanes 0-15 -> M=lane, halves[0..7]=K 0..7, halves[8..15]=K16..23
//                 lanes16-31 -> M=lane-16, halves=K 8..15 / 24..31
//  B (32x16 f16): lanes 0-15 -> N=lane, halves[0..7]=K 0..7, halves[8..15]=K16..23
//                 lanes16-31 -> N=lane-16, halves=K 8..15 / 24..31
// ---------------------------------------------------------------------------
__device__ __forceinline__ v16h load_afrag(const _Float16* lds, int mBase, int lane) {
    const int m  = lane & 15;
    const int hk = (lane >> 4) * 8;
    const _Float16* p = lds + (mBase + m) * LDH + hk;
    v16h a;
#pragma unroll
    for (int e = 0; e < 8; ++e) { a[e] = p[e]; a[e + 8] = p[e + 16]; }
    return a;
}

__device__ __forceinline__ v16h load_bfrag(const _Float16* lds, int nBase, int lane) {
    const int n  = lane & 15;
    const int hk = (lane >> 4) * 8;
    const _Float16* p = lds + (nBase + n) * LDH + hk;
    v8h lo = *(const v8h*)p;
    v8h hi = *(const v8h*)(p + 16);
    v16h b;
#pragma unroll
    for (int e = 0; e < 8; ++e) { b[e] = lo[e]; b[e + 8] = hi[e]; }
    return b;
}

// ---------------------------------------------------------------------------
// Kernel 1: C[m][n] = scale * sum_k A_f32[m][k] * Bw_f32[n][k]   (C = A·B^T)
// 256x128 block tile, 8 waves, each wave a 64x64 tile (16 WMMAs/k-step).
// Double-buffered LDS, register prefetch, one barrier per k-step.
// ---------------------------------------------------------------------------
__global__ __launch_bounds__(256) void gemm_f32_abT_f16(
    const float* __restrict__ A, const float* __restrict__ Bw,
    _Float16* __restrict__ C, int M, int N, int K, float scale)
{
    __shared__ _Float16 lA[2][256 * LDH];
    __shared__ _Float16 lB[2][128 * LDH];
    const int tid  = threadIdx.x;
    const int lane = tid & 31;
    const int wave = tid >> 5;
    const int m0 = blockIdx.y * 256;
    const int n0 = blockIdx.x * 128;
    const int wm = (wave & 3) * 64;
    const int wn = (wave >> 2) * 64;

    v8f acc[4][4];
#pragma unroll
    for (int i = 0; i < 4; ++i)
#pragma unroll
        for (int j = 0; j < 4; ++j)
#pragma unroll
            for (int e = 0; e < 8; ++e) acc[i][j][e] = 0.0f;

    // staging map: A -> thread stages full row tid (32 halves)
    //              B -> thread stages half row (tid>>1, 16 halves)
    const int br = tid >> 1, bc = (tid & 1) * 16;

    v8h aPre[4], bPre[2];

    auto loadGlobal = [&](int kt) {
        const float* sa = A + (size_t)(m0 + tid) * K + kt;
#pragma unroll
        for (int i = 0; i < 4; ++i) {
            float4 f0 = *(const float4*)(sa + i * 8);
            float4 f1 = *(const float4*)(sa + i * 8 + 4);
            aPre[i][0] = (_Float16)f0.x; aPre[i][1] = (_Float16)f0.y;
            aPre[i][2] = (_Float16)f0.z; aPre[i][3] = (_Float16)f0.w;
            aPre[i][4] = (_Float16)f1.x; aPre[i][5] = (_Float16)f1.y;
            aPre[i][6] = (_Float16)f1.z; aPre[i][7] = (_Float16)f1.w;
        }
        const float* sb = Bw + (size_t)(n0 + br) * K + kt + bc;
#pragma unroll
        for (int i = 0; i < 2; ++i) {
            float4 f0 = *(const float4*)(sb + i * 8);
            float4 f1 = *(const float4*)(sb + i * 8 + 4);
            bPre[i][0] = (_Float16)f0.x; bPre[i][1] = (_Float16)f0.y;
            bPre[i][2] = (_Float16)f0.z; bPre[i][3] = (_Float16)f0.w;
            bPre[i][4] = (_Float16)f1.x; bPre[i][5] = (_Float16)f1.y;
            bPre[i][6] = (_Float16)f1.z; bPre[i][7] = (_Float16)f1.w;
        }
    };
    auto storeLds = [&](int buf) {
        _Float16* da = &lA[buf][tid * LDH];
#pragma unroll
        for (int i = 0; i < 4; ++i) *(v8h*)(da + i * 8) = aPre[i];
        _Float16* db = &lB[buf][br * LDH + bc];
        *(v8h*)db       = bPre[0];
        *(v8h*)(db + 8) = bPre[1];
    };

    loadGlobal(0);
    storeLds(0);
    __syncthreads();

    const int nsteps = K / TK;
    for (int s = 0; s < nsteps; ++s) {
        const int cur = s & 1;
        if (s + 1 < nsteps) loadGlobal((s + 1) * TK);

        const _Float16* la = lA[cur];
        const _Float16* lb = lB[cur];
        v16h af[4], bf[4];
#pragma unroll
        for (int i = 0; i < 4; ++i) af[i] = load_afrag(la, wm + i * 16, lane);
#pragma unroll
        for (int j = 0; j < 4; ++j) bf[j] = load_bfrag(lb, wn + j * 16, lane);
#pragma unroll
        for (int i = 0; i < 4; ++i)
#pragma unroll
            for (int j = 0; j < 4; ++j)
                acc[i][j] = __builtin_amdgcn_wmma_f32_16x16x32_f16(
                    false, af[i], false, bf[j], (short)0, acc[i][j], false, false);

        if (s + 1 < nsteps) storeLds(cur ^ 1);
        __syncthreads();
    }

    // Epilogue: VGPR r, lanes0-15:(M=r,N=lane), lanes16-31:(M=r+8,N=lane-16)
    const int cn   = lane & 15;
    const int rsel = (lane >> 4) * 8;
#pragma unroll
    for (int i = 0; i < 4; ++i)
#pragma unroll
        for (int j = 0; j < 4; ++j)
#pragma unroll
            for (int r = 0; r < 8; ++r) {
                const int m = m0 + wm + i * 16 + rsel + r;
                const int n = n0 + wn + j * 16 + cn;
                C[(size_t)m * N + n] = (_Float16)(acc[i][j][r] * scale);
            }
}

// ---------------------------------------------------------------------------
// Kernel 2: scores = scale * Q·K^T   (f16 in, f16 out), batched via blockIdx.z
// ---------------------------------------------------------------------------
__global__ __launch_bounds__(256) void gemm_f16_abT_f16(
    const _Float16* __restrict__ Ab, const _Float16* __restrict__ Bb,
    _Float16* __restrict__ Cb, int M, int N, int K, float scale,
    long aBat, long bBat, long cBat)
{
    const _Float16* A  = Ab + (size_t)blockIdx.z * aBat;
    const _Float16* Bw = Bb + (size_t)blockIdx.z * bBat;
    _Float16*       C  = Cb + (size_t)blockIdx.z * cBat;

    __shared__ _Float16 lA[2][256 * LDH];
    __shared__ _Float16 lB[2][128 * LDH];
    const int tid  = threadIdx.x;
    const int lane = tid & 31;
    const int wave = tid >> 5;
    const int m0 = blockIdx.y * 256;
    const int n0 = blockIdx.x * 128;
    const int wm = (wave & 3) * 64;
    const int wn = (wave >> 2) * 64;

    v8f acc[4][4];
#pragma unroll
    for (int i = 0; i < 4; ++i)
#pragma unroll
        for (int j = 0; j < 4; ++j)
#pragma unroll
            for (int e = 0; e < 8; ++e) acc[i][j][e] = 0.0f;

    const int br = tid >> 1, bc = (tid & 1) * 16;
    v8h aPre[4], bPre[2];

    auto loadGlobal = [&](int kt) {
        const _Float16* sa = A + (size_t)(m0 + tid) * K + kt;
#pragma unroll
        for (int i = 0; i < 4; ++i) aPre[i] = *(const v8h*)(sa + i * 8);
        const _Float16* sb = Bw + (size_t)(n0 + br) * K + kt + bc;
        bPre[0] = *(const v8h*)sb;
        bPre[1] = *(const v8h*)(sb + 8);
    };
    auto storeLds = [&](int buf) {
        _Float16* da = &lA[buf][tid * LDH];
#pragma unroll
        for (int i = 0; i < 4; ++i) *(v8h*)(da + i * 8) = aPre[i];
        _Float16* db = &lB[buf][br * LDH + bc];
        *(v8h*)db       = bPre[0];
        *(v8h*)(db + 8) = bPre[1];
    };

    loadGlobal(0);
    storeLds(0);
    __syncthreads();

    const int nsteps = K / TK;
    for (int s = 0; s < nsteps; ++s) {
        const int cur = s & 1;
        if (s + 1 < nsteps) loadGlobal((s + 1) * TK);

        const _Float16* la = lA[cur];
        const _Float16* lb = lB[cur];
        v16h af[4], bf[4];
#pragma unroll
        for (int i = 0; i < 4; ++i) af[i] = load_afrag(la, wm + i * 16, lane);
#pragma unroll
        for (int j = 0; j < 4; ++j) bf[j] = load_bfrag(lb, wn + j * 16, lane);
#pragma unroll
        for (int i = 0; i < 4; ++i)
#pragma unroll
            for (int j = 0; j < 4; ++j)
                acc[i][j] = __builtin_amdgcn_wmma_f32_16x16x32_f16(
                    false, af[i], false, bf[j], (short)0, acc[i][j], false, false);

        if (s + 1 < nsteps) storeLds(cur ^ 1);
        __syncthreads();
    }

    const int cn   = lane & 15;
    const int rsel = (lane >> 4) * 8;
#pragma unroll
    for (int i = 0; i < 4; ++i)
#pragma unroll
        for (int j = 0; j < 4; ++j)
#pragma unroll
            for (int r = 0; r < 8; ++r) {
                const int m = m0 + wm + i * 16 + rsel + r;
                const int n = n0 + wn + j * 16 + cn;
                C[(size_t)m * N + n] = (_Float16)(acc[i][j][r] * scale);
            }
}

// ---------------------------------------------------------------------------
// Kernel 3: in-place row softmax, f16, ncols == 2048, 256 threads (8/thread).
// Single global read + single global write; all passes in registers.
// ---------------------------------------------------------------------------
__global__ __launch_bounds__(256) void softmax_rows(_Float16* __restrict__ S, int ncols)
{
    __shared__ float red[256];
    _Float16* p = S + (size_t)blockIdx.x * ncols;
    const int t = threadIdx.x;

    v8h d = *(const v8h*)(p + t * 8);
    float f[8];
#pragma unroll
    for (int i = 0; i < 8; ++i) f[i] = (float)d[i];

    float lmax = f[0];
#pragma unroll
    for (int i = 1; i < 8; ++i) lmax = fmaxf(lmax, f[i]);
    red[t] = lmax; __syncthreads();
    for (int s = 128; s > 0; s >>= 1) {
        if (t < s) red[t] = fmaxf(red[t], red[t + s]);
        __syncthreads();
    }
    const float m = red[0]; __syncthreads();

    float lsum = 0.0f;
#pragma unroll
    for (int i = 0; i < 8; ++i) { f[i] = __expf(f[i] - m); lsum += f[i]; }
    red[t] = lsum; __syncthreads();
    for (int s = 128; s > 0; s >>= 1) {
        if (t < s) red[t] += red[t + s];
        __syncthreads();
    }
    const float inv = 1.0f / red[0];

    v8h o;
#pragma unroll
    for (int i = 0; i < 8; ++i) o[i] = (_Float16)(f[i] * inv);
    *(v8h*)(p + t * 8) = o;
}

// ---------------------------------------------------------------------------
// Kernel 4: O = P·V  (P:[M][K] f16 row-major, V:[K][N] f16 row-major, O f32)
// V tile staged TRANSPOSED into LDS so B-fragment loads stay contiguous.
// ---------------------------------------------------------------------------
__global__ __launch_bounds__(256) void gemm_f16_pv_f32(
    const _Float16* __restrict__ Pb, const _Float16* __restrict__ Vb,
    float* __restrict__ Ob, int M, int N, int K,
    long pBat, long vBat, long oBat)
{
    const _Float16* P = Pb + (size_t)blockIdx.z * pBat;
    const _Float16* V = Vb + (size_t)blockIdx.z * vBat;
    float*          O = Ob + (size_t)blockIdx.z * oBat;

    __shared__ _Float16 lA[2][256 * LDH];
    __shared__ _Float16 lB[2][128 * LDH];   // lB[n][k] (transposed V tile)
    const int tid  = threadIdx.x;
    const int lane = tid & 31;
    const int wave = tid >> 5;
    const int m0 = blockIdx.y * 256;
    const int n0 = blockIdx.x * 128;
    const int wm = (wave & 3) * 64;
    const int wn = (wave >> 2) * 64;

    v8f acc[4][4];
#pragma unroll
    for (int i = 0; i < 4; ++i)
#pragma unroll
        for (int j = 0; j < 4; ++j)
#pragma unroll
            for (int e = 0; e < 8; ++e) acc[i][j][e] = 0.0f;

    const int vk = tid >> 3;          // 0..31  key index within k-chunk
    const int vn = (tid & 7) * 16;    // 0..112 n sub-offset
    v8h aPre[4], bPre[2];

    auto loadGlobal = [&](int kt) {
        const _Float16* sa = P + (size_t)(m0 + tid) * K + kt;
#pragma unroll
        for (int i = 0; i < 4; ++i) aPre[i] = *(const v8h*)(sa + i * 8);
        const _Float16* sb = V + (size_t)(kt + vk) * N + n0 + vn;
        bPre[0] = *(const v8h*)sb;
        bPre[1] = *(const v8h*)(sb + 8);
    };
    auto storeLds = [&](int buf) {
        _Float16* da = &lA[buf][tid * LDH];
#pragma unroll
        for (int i = 0; i < 4; ++i) *(v8h*)(da + i * 8) = aPre[i];
        _Float16* db = &lB[buf][0];
#pragma unroll
        for (int i = 0; i < 8; ++i) {
            db[(vn + i)     * LDH + vk] = bPre[0][i];
            db[(vn + 8 + i) * LDH + vk] = bPre[1][i];
        }
    };

    loadGlobal(0);
    storeLds(0);
    __syncthreads();

    const int nsteps = K / TK;
    for (int s = 0; s < nsteps; ++s) {
        const int cur = s & 1;
        if (s + 1 < nsteps) loadGlobal((s + 1) * TK);

        const _Float16* la = lA[cur];
        const _Float16* lb = lB[cur];
        v16h af[4], bf[4];
#pragma unroll
        for (int i = 0; i < 4; ++i) af[i] = load_afrag(la, wm + i * 16, lane);
#pragma unroll
        for (int j = 0; j < 4; ++j) bf[j] = load_bfrag(lb, wn + j * 16, lane);
#pragma unroll
        for (int i = 0; i < 4; ++i)
#pragma unroll
            for (int j = 0; j < 4; ++j)
                acc[i][j] = __builtin_amdgcn_wmma_f32_16x16x32_f16(
                    false, af[i], false, bf[j], (short)0, acc[i][j], false, false);

        if (s + 1 < nsteps) storeLds(cur ^ 1);
        __syncthreads();
    }

    const int cn   = lane & 15;
    const int rsel = (lane >> 4) * 8;
#pragma unroll
    for (int i = 0; i < 4; ++i)
#pragma unroll
        for (int j = 0; j < 4; ++j)
#pragma unroll
            for (int r = 0; r < 8; ++r) {
                const int m = m0 + wm + i * 16 + rsel + r;
                const int n = n0 + wn + j * 16 + cn;
                O[(size_t)m * N + n] = acc[i][j][r];
            }
}

// ---------------------------------------------------------------------------
extern "C" void kernel_launch(void* const* d_in, const int* in_sizes, int n_in,
                              void* d_out, int out_size, void* d_ws, size_t ws_size,
                              hipStream_t stream) {
    constexpr int Bn = 4, S = 2048, D = 1024;
    const float* x  = (const float*)d_in[0];
    const float* Wq = (const float*)d_in[1];
    const float* Wk = (const float*)d_in[2];
    const float* Wv = (const float*)d_in[3];
    float* out = (float*)d_out;

    const size_t MK = (size_t)Bn * S * D;          // 8,388,608 elems
    _Float16* Qh = (_Float16*)d_ws;                // [B*S][D] f16
    _Float16* Kh = Qh + MK;
    _Float16* Vh = Kh + MK;
    _Float16* Sc = Vh + MK;                        // [B][S][S] f16 scores/probs

    const dim3 blk(256);
    const int Mrows = Bn * S;                      // 8192

    // 1) QKV projections: y = x @ W^T  (f32 -> f16)
    const dim3 g1(D / 128, Mrows / 256);           // 8 x 32
    gemm_f32_abT_f16<<<g1, blk, 0, stream>>>(x, Wq, Qh, Mrows, D, D, 1.0f);
    gemm_f32_abT_f16<<<g1, blk, 0, stream>>>(x, Wk, Kh, Mrows, D, D, 1.0f);
    gemm_f32_abT_f16<<<g1, blk, 0, stream>>>(x, Wv, Vh, Mrows, D, D, 1.0f);

    // 2) scores = (Q @ K^T) / sqrt(D)   per batch
    const dim3 g2(S / 128, S / 256, Bn);           // 16 x 8 x 4
    gemm_f16_abT_f16<<<g2, blk, 0, stream>>>(
        Qh, Kh, Sc, S, S, D, 0.03125f,
        (long)S * D, (long)S * D, (long)S * S);

    // 3) softmax over last axis, in place
    softmax_rows<<<dim3(Bn * S), blk, 0, stream>>>(Sc, S);

    // 4) out = P @ V   per batch (f32 output)
    const dim3 g4(D / 128, S / 256, Bn);           // 8 x 8 x 4
    gemm_f16_pv_f32<<<g4, blk, 0, stream>>>(
        Sc, Vh, out, S, D, S,
        (long)S * S, (long)S * D, (long)S * D);
}
